// HeteroLayer_11252814315837
// MI455X (gfx1250) — matrix operand order repaired and checked by reading, not compiled, not run on hardware
//
#include <hip/hip_runtime.h>
#include <math.h>

// ---------------------------------------------------------------------------
// HeteroGNN layer for MI455X (gfx1250, wave32, WMMA).
// GEMMs via V_WMMA_F32_16X16X32_BF16 with split-bf16 (hi/lo) operands for
// near-fp32 accuracy; edge aggregation via global_atomic_add_f32.
// Each wave computes a 32(M)x64(N) strip: two A fragments share every B
// fragment -> 24 WMMAs per 24 b128 loads per k-step.
// ---------------------------------------------------------------------------

typedef __attribute__((ext_vector_type(16))) __bf16 v16bf;
typedef __attribute__((ext_vector_type(8)))  float  v8f;

#define D 256   // D_IN == D_OUT == 256

// ---- weight prep: W f32 [k][n] -> transposed bf16 hi/lo [n][k] -------------
// Output layout per matrix m: wb + m*131072 : 65536 bf16 hi, then 65536 bf16 lo.
__global__ void prep_weights_kernel(const float* __restrict__ w0,
                                    const float* __restrict__ w1,
                                    const float* __restrict__ w2,
                                    const float* __restrict__ w3,
                                    const float* __restrict__ w4,
                                    __bf16* __restrict__ wb) {
    int idx = blockIdx.x * blockDim.x + threadIdx.x;    // 5 * 65536 threads
    int mat = idx >> 16;
    int rem = idx & 65535;
    int k = rem >> 8, n = rem & 255;
    const float* W = (mat == 0) ? w0 : (mat == 1) ? w1 : (mat == 2) ? w2
                     : (mat == 3) ? w3 : w4;
    float v   = W[k * D + n];
    __bf16 h  = (__bf16)v;
    __bf16 l  = (__bf16)(v - (float)h);
    __bf16* whi = wb + (size_t)mat * 131072;
    whi[n * D + k]         = h;
    whi[65536 + n * D + k] = l;
}

// Load 16 f32 of one A-fragment row chunk and split into bf16 hi/lo.
// 16-bit A 16x32 layout (contiguous, matching B): elems i -> k = half*16 + i.
__device__ __forceinline__ void load_split_a(const float* __restrict__ p,
                                             v16bf& h, v16bf& l) {
    float4 f0 = *(const float4*)(p);
    float4 f1 = *(const float4*)(p + 4);
    float4 f2 = *(const float4*)(p + 8);
    float4 f3 = *(const float4*)(p + 12);
    float af[16] = {f0.x, f0.y, f0.z, f0.w, f1.x, f1.y, f1.z, f1.w,
                    f2.x, f2.y, f2.z, f2.w, f3.x, f3.y, f3.z, f3.w};
    #pragma unroll
    for (int i = 0; i < 16; ++i) {
        __bf16 hi = (__bf16)af[i];
        h[i] = hi;
        l[i] = (__bf16)(af[i] - (float)hi);
    }
}

// ---- GEMM: Out[m][n] = sum_k X[m][k]*W[k][n] + bias[n] ---------------------
// Block: 256 thr = 8 waves. Wave w -> rows [blk.x*256 + w*32, +32),
// cols [blk.y*64, +64). nrows must be a multiple of 16.
__global__ __launch_bounds__(256)
void gemm_bf16x3_kernel(const float* __restrict__ X,
                        const __bf16* __restrict__ Wb,   // hi at 0, lo at +65536
                        const float* __restrict__ bias,
                        float* __restrict__ Out, int nrows) {
    const int lane = threadIdx.x & 31;
    const int wave = threadIdx.x >> 5;
    const int m0   = blockIdx.x * 256 + wave * 32;
    if (m0 >= nrows) return;                 // wave-uniform: EXEC stays all-1s
    const bool has2 = (m0 + 16) < nrows;     // wave-uniform
    const int  mB   = has2 ? m0 + 16 : m0;   // clamp: reads stay in-bounds
    const int n0   = blockIdx.y * 64;
    const int half = lane >> 4;              // 0 or 1 (lane group)
    const int lrow = lane & 15;

    const float* xrow0 = X + (size_t)(m0 + lrow) * D + half * 16;
    const float* xrow1 = X + (size_t)(mB + lrow) * D + half * 16;

    v8f acc0[4] = {};
    v8f acc1[4] = {};

    for (int kk = 0; kk < D; kk += 32) {
        v16bf a0h, a0l, a1h, a1l;
        load_split_a(xrow0 + kk, a0h, a0l);
        load_split_a(xrow1 + kk, a1h, a1l);
        #pragma unroll
        for (int t = 0; t < 4; ++t) {
            // B fragment: 16-bit B 32x16, lane -> col n0+t*16+lrow,
            // elems i -> k = half*16 + i (contiguous in transposed W)
            const __bf16* bp =
                Wb + (size_t)(n0 + t * 16 + lrow) * D + half * 16 + kk;
            v16bf bh = *(const v16bf*)bp;
            v16bf bl = *(const v16bf*)(bp + 65536);
            acc0[t] = __builtin_amdgcn_wmma_f32_16x16x32_bf16(
                false, a0h, false, bh, (short)0, acc0[t], false, false);
            acc0[t] = __builtin_amdgcn_wmma_f32_16x16x32_bf16(
                false, a0l, false, bh, (short)0, acc0[t], false, false);
            acc0[t] = __builtin_amdgcn_wmma_f32_16x16x32_bf16(
                false, a0h, false, bl, (short)0, acc0[t], false, false);
            acc1[t] = __builtin_amdgcn_wmma_f32_16x16x32_bf16(
                false, a1h, false, bh, (short)0, acc1[t], false, false);
            acc1[t] = __builtin_amdgcn_wmma_f32_16x16x32_bf16(
                false, a1l, false, bh, (short)0, acc1[t], false, false);
            acc1[t] = __builtin_amdgcn_wmma_f32_16x16x32_bf16(
                false, a1h, false, bl, (short)0, acc1[t], false, false);
        }
    }

    // C/D layout: lanes 0-15 -> M=j, lanes 16-31 -> M=8+j; N = lane&15.
    const int mr0 = m0 + half * 8;
    const int mr1 = mB + half * 8;
    #pragma unroll
    for (int t = 0; t < 4; ++t) {
        int col  = n0 + t * 16 + lrow;
        float bv = bias[col];
        #pragma unroll
        for (int j = 0; j < 8; ++j)
            Out[(size_t)(mr0 + j) * D + col] = acc0[t][j] + bv;
        if (has2) {
            #pragma unroll
            for (int j = 0; j < 8; ++j)
                Out[(size_t)(mr1 + j) * D + col] = acc1[t][j] + bv;
        }
    }
}

// ---- edge scatter: out[dst[e]] += msg[src[e]] * (ew ? ew[e] : 1) -----------
// One wave per edge, 8 floats per lane, hardware f32 atomics into L2.
__global__ void scatter_add_kernel(const float* __restrict__ msg,
                                   const int* __restrict__ src,
                                   const int* __restrict__ dst,
                                   const float* __restrict__ ew,
                                   float* __restrict__ out, int E) {
    int t = blockIdx.x * blockDim.x + threadIdx.x;
    int e = t >> 5;
    if (e >= E) return;
    int lane = t & 31;
    float scale = ew ? ew[e] : 1.0f;
    const float* m = msg + (size_t)src[e] * D + lane * 8;
    float*       o = out + (size_t)dst[e] * D + lane * 8;
    float4 a = *(const float4*)m;
    float4 b = *(const float4*)(m + 4);
    unsafeAtomicAdd(o + 0, a.x * scale);
    unsafeAtomicAdd(o + 1, a.y * scale);
    unsafeAtomicAdd(o + 2, a.z * scale);
    unsafeAtomicAdd(o + 3, a.w * scale);
    unsafeAtomicAdd(o + 4, b.x * scale);
    unsafeAtomicAdd(o + 5, b.y * scale);
    unsafeAtomicAdd(o + 6, b.z * scale);
    unsafeAtomicAdd(o + 7, b.w * scale);
}

// ---- in-place ELU ----------------------------------------------------------
__global__ void elu_kernel(float4* __restrict__ p, int n4) {
    int i = blockIdx.x * blockDim.x + threadIdx.x;
    if (i >= n4) return;
    float4 v = p[i];
    v.x = v.x > 0.0f ? v.x : (expf(v.x) - 1.0f);
    v.y = v.y > 0.0f ? v.y : (expf(v.y) - 1.0f);
    v.z = v.z > 0.0f ? v.z : (expf(v.z) - 1.0f);
    v.w = v.w > 0.0f ? v.w : (expf(v.w) - 1.0f);
    p[i] = v;
}

extern "C" void kernel_launch(void* const* d_in, const int* in_sizes, int n_in,
                              void* d_out, int out_size, void* d_ws, size_t ws_size,
                              hipStream_t stream) {
    const float* x_paper   = (const float*)d_in[0];
    const float* x_author  = (const float*)d_in[1];
    const int*   cites_src = (const int*)d_in[2];
    const int*   cites_dst = (const int*)d_in[3];
    const float* cites_w   = (const float*)d_in[4];
    const int*   writes_src  = (const int*)d_in[5];
    const int*   writes_dst  = (const int*)d_in[6];
    const int*   written_src = (const int*)d_in[7];
    const int*   written_dst = (const int*)d_in[8];
    const float* W_self_p  = (const float*)d_in[9];
    const float* b_self_p  = (const float*)d_in[10];
    const float* W_self_a  = (const float*)d_in[11];
    const float* b_self_a  = (const float*)d_in[12];
    const float* W_cites   = (const float*)d_in[13];
    const float* b_cites   = (const float*)d_in[14];
    const float* W_writes  = (const float*)d_in[15];
    const float* b_writes  = (const float*)d_in[16];
    const float* W_written = (const float*)d_in[17];
    const float* b_written = (const float*)d_in[18];

    const int NP = in_sizes[0] / D;     // 100000 papers
    const int NA = in_sizes[1] / D;     // 50000 authors
    const int EC = in_sizes[2];         // 300000 cites edges
    const int EW = in_sizes[5];         // 200000 writes edges
    const int EB = in_sizes[7];         // 200000 written edges

    float* out_p = (float*)d_out;
    float* out_a = out_p + (size_t)NP * D;

    // workspace: [msg buffer NP*256 f32][5 x (hi+lo) bf16 weights]
    float*  msg = (float*)d_ws;
    __bf16* wb  = (__bf16*)((char*)d_ws + (size_t)NP * D * sizeof(float));

    // 1) convert weights to transposed split-bf16 (order: self_p, self_a,
    //    cites, writes, written)
    prep_weights_kernel<<<(5 * 65536) / 256, 256, 0, stream>>>(
        W_self_p, W_self_a, W_cites, W_writes, W_written, wb);

    dim3 blk(256);
    dim3 gP((NP + 255) / 256, 4);
    dim3 gA((NA + 255) / 256, 4);

    // 2) self terms (overwrite d_out -> idempotent across replays)
    gemm_bf16x3_kernel<<<gP, blk, 0, stream>>>(x_paper,  wb + 0 * 131072,
                                               b_self_p, out_p, NP);
    gemm_bf16x3_kernel<<<gA, blk, 0, stream>>>(x_author, wb + 1 * 131072,
                                               b_self_a, out_a, NA);

    // 3) cites: msg = x_paper @ W_cites + b ; out_p += seg_sum(msg[src]*w)
    gemm_bf16x3_kernel<<<gP, blk, 0, stream>>>(x_paper, wb + 2 * 131072,
                                               b_cites, msg, NP);
    scatter_add_kernel<<<(unsigned)(((size_t)EC * 32 + 255) / 256), blk, 0, stream>>>(
        msg, cites_src, cites_dst, cites_w, out_p, EC);

    // 4) writes: msg = x_author @ W_writes + b ; out_p += seg_sum(msg[src])
    gemm_bf16x3_kernel<<<gA, blk, 0, stream>>>(x_author, wb + 3 * 131072,
                                               b_writes, msg, NA);
    scatter_add_kernel<<<(unsigned)(((size_t)EW * 32 + 255) / 256), blk, 0, stream>>>(
        msg, writes_src, writes_dst, nullptr, out_p, EW);

    // 5) written: msg = x_paper @ W_written + b ; out_a += seg_sum(msg[src])
    gemm_bf16x3_kernel<<<gP, blk, 0, stream>>>(x_paper, wb + 4 * 131072,
                                               b_written, msg, NP);
    scatter_add_kernel<<<(unsigned)(((size_t)EB * 32 + 255) / 256), blk, 0, stream>>>(
        msg, written_src, written_dst, nullptr, out_a, EB);

    // 6) ELU over both outputs in place
    int n4 = (NP + NA) * (D / 4);
    elu_kernel<<<(n4 + 255) / 256, blk, 0, stream>>>((float4*)d_out, n4);
}
